// MultiHeadLatentAttention_4999341932992
// MI455X (gfx1250) — matrix operand order repaired and checked by reading, not compiled, not run on hardware
//
#include <hip/hip_runtime.h>
#include <hip/hip_bf16.h>

// ---------------------------------------------------------------------------
// MLA forward for MI455X (gfx1250, wave32, WMMA).
// All GEMM-shaped work (projections, QK^T, P*V, out-proj) runs on
// v_wmma_f32_16x16x32_bf16. Precision: bf16 operands / fp32 accumulate.
// GEMM is software-pipelined with ping/pong fragment buffers held in
// *named variables* (compile-time indices only) so nothing is demoted to
// scratch/LDS by PromoteAlloca.
// ---------------------------------------------------------------------------

typedef __attribute__((ext_vector_type(16))) __bf16 bf16x16;
typedef __attribute__((ext_vector_type(8)))  __bf16 bf16x8;
typedef __attribute__((ext_vector_type(8)))  float  f32x8;

union Frag16 { bf16x16 v; bf16x8 h[2]; };

__device__ __forceinline__ f32x8 wmma_bf16(bf16x16 a, bf16x16 b, f32x8 c) {
  // (neg_a, A, neg_b, B, c_mod, C, reuse_a, reuse_b)
  return __builtin_amdgcn_wmma_f32_16x16x32_bf16(false, a, false, b, (short)0, c, false, false);
}

__device__ __forceinline__ f32x8 zero_f32x8() {
  f32x8 z = {0.f, 0.f, 0.f, 0.f, 0.f, 0.f, 0.f, 0.f};
  return z;
}

// Problem constants (from reference)
#define BB 2
#define SS 2048
#define HH 2048
#define NHH 16
#define HDD 128
#define LL 512
#define MM (BB * SS)   // 4096 flattened tokens

// ---------------------------------------------------------------------------
// fp32 -> bf16 conversion (grid-stride)
// ---------------------------------------------------------------------------
__global__ void f32_to_bf16(const float* __restrict__ x, __bf16* __restrict__ y, int n) {
  int i = blockIdx.x * blockDim.x + threadIdx.x;
  int stride = gridDim.x * blockDim.x;
  for (; i < n; i += stride) y[i] = (__bf16)x[i];
}

// ---------------------------------------------------------------------------
// Tiled WMMA GEMM: Y = X(M,K) * W(N,K)^T + bias.
// Block = 256 threads = 8 waves; wave w owns a 32(M) x 64(N) tile
// (8 f32 accumulators); block tile = 256 x 64. K stepped by 32; software
// pipeline manually unrolled x2 (ping/pong named buffers -> registers only).
// Requires K % 64 == 0 (all call sites: 512 or 2048).
//
// Fragment addressing uses contiguous k order (k = 16h + s), the same for A and B:
//   A (16x32): lane m=lane&15; 16 halves K = k0 + (lane>>4)*16 .. +15
//              -> two 16B loads.
//   B (32x16): lane n=lane&15; 16 halves K = k0 + (lane>>4)*16 .. +15
//              -> one 32B load (W row-major over K == B column).
//   C (16x16 f32): vgpr r: M = r + 8*(lane>>4), N = lane&15.
//
// mode 0: bf16 store Y[m*ldy + col0 + n]
// mode 1: f32  store Y[m*ldy + col0 + n]
// mode 2: bf16 store transposed-per-head for V:  vt[((b*16+h)*128+d)*2048 + s]
// ---------------------------------------------------------------------------
__global__ __launch_bounds__(256)
void wmma_gemm_xwt(const __bf16* __restrict__ X, const __bf16* __restrict__ W,
                   const float* __restrict__ bias, void* __restrict__ Y,
                   int M, int N, int K, int ldy, int col0, int mode) {
  const int lane = threadIdx.x & 31;
  const int wave = threadIdx.x >> 5;
  const int m_base = blockIdx.y * 256 + wave * 32;
  const int n_base = blockIdx.x * 64;
  const int lm = lane & 15;
  const int lh = lane >> 4;

  f32x8 acc00 = zero_f32x8(), acc01 = zero_f32x8(), acc02 = zero_f32x8(), acc03 = zero_f32x8();
  f32x8 acc10 = zero_f32x8(), acc11 = zero_f32x8(), acc12 = zero_f32x8(), acc13 = zero_f32x8();

  const __bf16* arow0 = X + (size_t)(m_base + lm) * K + (lh << 4);
  const __bf16* arow1 = arow0 + (size_t)16 * K;
  const __bf16* brow0 = W + (size_t)(n_base + 0 * 16 + lm) * K + (lh << 4);
  const __bf16* brow1 = W + (size_t)(n_base + 1 * 16 + lm) * K + (lh << 4);
  const __bf16* brow2 = W + (size_t)(n_base + 2 * 16 + lm) * K + (lh << 4);
  const __bf16* brow3 = W + (size_t)(n_base + 3 * 16 + lm) * K + (lh << 4);

  // ping/pong slabs as named variables (never dynamically indexed)
  Frag16 pA0, pA1;          bf16x16 pB0, pB1, pB2, pB3;   // ping
  Frag16 qA0, qA1;          bf16x16 qB0, qB1, qB2, qB3;   // pong

#define LOAD_SLAB(A0, A1, B0, B1, B2, B3, kk)                         \
  do {                                                                \
    A0.h[0] = *(const bf16x8*)(arow0 + (kk));                         \
    A0.h[1] = *(const bf16x8*)(arow0 + (kk) + 8);                     \
    A1.h[0] = *(const bf16x8*)(arow1 + (kk));                         \
    A1.h[1] = *(const bf16x8*)(arow1 + (kk) + 8);                     \
    B0 = *(const bf16x16*)(brow0 + (kk));                             \
    B1 = *(const bf16x16*)(brow1 + (kk));                             \
    B2 = *(const bf16x16*)(brow2 + (kk));                             \
    B3 = *(const bf16x16*)(brow3 + (kk));                             \
  } while (0)

#define COMPUTE_SLAB(A0, A1, B0, B1, B2, B3)                          \
  do {                                                                \
    acc00 = wmma_bf16(A0.v, B0, acc00);                               \
    acc01 = wmma_bf16(A0.v, B1, acc01);                               \
    acc02 = wmma_bf16(A0.v, B2, acc02);                               \
    acc03 = wmma_bf16(A0.v, B3, acc03);                               \
    acc10 = wmma_bf16(A1.v, B0, acc10);                               \
    acc11 = wmma_bf16(A1.v, B1, acc11);                               \
    acc12 = wmma_bf16(A1.v, B2, acc12);                               \
    acc13 = wmma_bf16(A1.v, B3, acc13);                               \
  } while (0)

  LOAD_SLAB(pA0, pA1, pB0, pB1, pB2, pB3, 0);
  int k0 = 0;
  for (; k0 + 64 < K; k0 += 64) {
    LOAD_SLAB(qA0, qA1, qB0, qB1, qB2, qB3, k0 + 32);
    if (k0 + 128 < K) {
      __builtin_prefetch(arow0 + k0 + 128, 0, 0);   // global_prefetch_b8
      __builtin_prefetch(arow1 + k0 + 128, 0, 0);
    }
    COMPUTE_SLAB(pA0, pA1, pB0, pB1, pB2, pB3);
    LOAD_SLAB(pA0, pA1, pB0, pB1, pB2, pB3, k0 + 64);
    COMPUTE_SLAB(qA0, qA1, qB0, qB1, qB2, qB3);
  }
  // tail: slabs k0 (in ping) and k0+32 (K % 64 == 0)
  LOAD_SLAB(qA0, qA1, qB0, qB1, qB2, qB3, k0 + 32);
  COMPUTE_SLAB(pA0, pA1, pB0, pB1, pB2, pB3);
  COMPUTE_SLAB(qA0, qA1, qB0, qB1, qB2, qB3);

#undef LOAD_SLAB
#undef COMPUTE_SLAB

  // ---- epilogue: bias + store ----
  f32x8 accv[2][4] = {{acc00, acc01, acc02, acc03}, {acc10, acc11, acc12, acc13}};
#pragma unroll
  for (int t = 0; t < 4; ++t) {
    const int n = n_base + t * 16 + lm;
    const float bv = bias ? bias[n] : 0.f;
#pragma unroll
    for (int mt = 0; mt < 2; ++mt) {
#pragma unroll
      for (int r = 0; r < 8; ++r) {
        const int m = m_base + mt * 16 + r + (lh << 3);
        const float val = accv[mt][t][r] + bv;
        if (mode == 0) {
          ((__bf16*)Y)[(size_t)m * ldy + col0 + n] = (__bf16)val;
        } else if (mode == 1) {
          ((float*)Y)[(size_t)m * ldy + col0 + n] = val;
        } else {
          const int bb = m >> 11, s = m & (SS - 1);
          const int h = n >> 7, d = n & (HDD - 1);
          ((__bf16*)Y)[((size_t)((bb * NHH + h) * HDD + d)) * SS + s] = (__bf16)val;
        }
      }
    }
  }
}

// ---------------------------------------------------------------------------
// RoPE: k_rope_raw (MM x 1024, bf16) -> rotated into km columns [1024,2048).
// ---------------------------------------------------------------------------
__global__ void rope_kernel(const __bf16* __restrict__ kr, __bf16* __restrict__ kmat) {
  const int idx = blockIdx.x * blockDim.x + threadIdx.x;  // MM * 512 threads
  if (idx >= MM * 512) return;
  const int j = idx & 511;
  const int m = idx >> 9;
  const int pos = m & (SS - 1);
  // inv_freq = 10000^(-2j/1024) = exp(-2j/1024 * ln(10000))
  const float inv_freq = __expf(-(float)(2 * j) * (9.210340371976184f / 1024.f));
  const float ang = (float)pos * inv_freq;
  float sn, c;
  __sincosf(ang, &sn, &c);
  const float x1 = (float)kr[(size_t)m * 1024 + j];
  const float x2 = (float)kr[(size_t)m * 1024 + 512 + j];
  kmat[(size_t)m * HH + 1024 + j]       = (__bf16)(x1 * c - x2 * sn);
  kmat[(size_t)m * HH + 1024 + 512 + j] = (__bf16)(x1 * sn + x2 * c);
}

// ---------------------------------------------------------------------------
// Flash attention: one wave per (batch, head, 16-query tile).
// Q (16x128) in registers; keys streamed 32 at a time:
//   8 WMMA (QK^T) -> online softmax (half-wave shfl reductions, layout-exact)
//   -> exp'd P staged via per-wave LDS tile -> 8 WMMA (P*V).
// V is pre-transposed (vt[(b,h,d), s]) so P*V B-fragments load contiguously.
// ---------------------------------------------------------------------------
__global__ __launch_bounds__(256)
void mla_flash_attn(const __bf16* __restrict__ q, const __bf16* __restrict__ k,
                    const __bf16* __restrict__ vt, __bf16* __restrict__ out) {
  __shared__ __bf16 pbuf[8][16][32];  // per-wave 16x32 exp'd score tile

  const int lane = threadIdx.x & 31;
  const int wave = threadIdx.x >> 5;
  const int task = blockIdx.x * 8 + wave;      // 0 .. 4095
  const int qt = task & 127;                   // query tile (S/16)
  const int h  = (task >> 7) & (NHH - 1);
  const int b  = task >> 11;
  const int lm = lane & 15;
  const int lh = lane >> 4;

  const __bf16* qbase = q + ((size_t)b * SS) * HH + h * HDD;
  const __bf16* kbase = k + ((size_t)b * SS) * HH + h * HDD;
  const __bf16* vbase = vt + ((size_t)(b * NHH + h) * HDD) * SS;

  // Q fragments (16 queries x 128 dims = 4 A-fragments), loaded once.
  Frag16 aQ[4];
  const __bf16* qrow = qbase + (size_t)(qt * 16 + lm) * HH + (lh << 4);
#pragma unroll
  for (int kd = 0; kd < 4; ++kd) {
    aQ[kd].h[0] = *(const bf16x8*)(qrow + kd * 32);
    aQ[kd].h[1] = *(const bf16x8*)(qrow + kd * 32 + 8);
  }

  float mrow[8], lrow[8];
  f32x8 accO[8];
#pragma unroll
  for (int r = 0; r < 8; ++r) { mrow[r] = -1e30f; lrow[r] = 0.f; }
#pragma unroll
  for (int dt = 0; dt < 8; ++dt) accO[dt] = zero_f32x8();

  const float scale = 0.08838834764831845f;  // 1/sqrt(128)

  for (int n0 = 0; n0 < SS; n0 += 32) {
    // ---- scores for 32 keys: two 16x16 tiles ----
    f32x8 s[2];
#pragma unroll
    for (int g = 0; g < 2; ++g) {
      f32x8 sc = zero_f32x8();
      const __bf16* krow = kbase + (size_t)(n0 + g * 16 + lm) * HH + (lh << 4);
#pragma unroll
      for (int kd = 0; kd < 4; ++kd) {
        bf16x16 bK = *(const bf16x16*)(krow + kd * 32);
        sc = wmma_bf16(aQ[kd].v, bK, sc);
      }
      s[g] = sc;
    }

    // ---- online softmax (row stats replicated across each 16-lane half) ----
    float pr0[8], pr1[8], sf[8];
#pragma unroll
    for (int r = 0; r < 8; ++r) {
      const float a0 = s[0][r] * scale;
      const float a1 = s[1][r] * scale;
      float cm = fmaxf(a0, a1);
#pragma unroll
      for (int msk = 1; msk < 16; msk <<= 1)
        cm = fmaxf(cm, __shfl_xor(cm, msk, 32));
      const float nm = fmaxf(mrow[r], cm);
      sf[r] = __expf(mrow[r] - nm);
      mrow[r] = nm;
      pr0[r] = __expf(a0 - nm);
      pr1[r] = __expf(a1 - nm);
      float rs = pr0[r] + pr1[r];
#pragma unroll
      for (int msk = 1; msk < 16; msk <<= 1)
        rs += __shfl_xor(rs, msk, 32);
      lrow[r] = lrow[r] * sf[r] + rs;
    }
#pragma unroll
    for (int dt = 0; dt < 8; ++dt)
#pragma unroll
      for (int r = 0; r < 8; ++r) accO[dt][r] *= sf[r];

    // ---- stage P (C-layout -> A-layout) through LDS, wave-synchronous ----
#pragma unroll
    for (int r = 0; r < 8; ++r) {
      const int row = r + (lh << 3);
      pbuf[wave][row][lm]      = (__bf16)pr0[r];
      pbuf[wave][row][16 + lm] = (__bf16)pr1[r];
    }
    __asm__ volatile("s_wait_dscnt 0" ::: "memory");  // stores visible to whole wave

    Frag16 aP;
    aP.h[0] = *(const bf16x8*)&pbuf[wave][lm][lh << 4];
    aP.h[1] = *(const bf16x8*)&pbuf[wave][lm][(lh << 4) + 8];

    // ---- accO += P(16x32) * V(32x128): 8 WMMA over d-tiles ----
#pragma unroll
    for (int dt = 0; dt < 8; ++dt) {
      const __bf16* vrow = vbase + (size_t)(dt * 16 + lm) * SS + n0 + (lh << 4);
      bf16x16 bV = *(const bf16x16*)vrow;
      accO[dt] = wmma_bf16(aP.v, bV, accO[dt]);
    }
    __asm__ volatile("s_wait_dscnt 0" ::: "memory");  // P reads done before overwrite
  }

  // ---- normalize and write head output (merged layout: row=token, col=h*128+d)
  const size_t orow0 = (size_t)b * SS + qt * 16;
#pragma unroll
  for (int dt = 0; dt < 8; ++dt) {
    const int col = h * HDD + dt * 16 + lm;
#pragma unroll
    for (int r = 0; r < 8; ++r) {
      const int m = r + (lh << 3);
      const float val = accO[dt][r] / lrow[r];
      out[(orow0 + m) * HH + col] = (__bf16)val;
    }
  }
}

// ---------------------------------------------------------------------------
// Host-side orchestration
// ---------------------------------------------------------------------------
extern "C" void kernel_launch(void* const* d_in, const int* in_sizes, int n_in,
                              void* d_out, int out_size, void* d_ws, size_t ws_size,
                              hipStream_t stream) {
  const float* hs       = (const float*)d_in[0];
  const float* Wq_down  = (const float*)d_in[1];
  const float* bq_down  = (const float*)d_in[2];
  const float* Wkv_down = (const float*)d_in[3];
  const float* bkv_down = (const float*)d_in[4];
  const float* Wq_up    = (const float*)d_in[5];
  const float* bq_up    = (const float*)d_in[6];
  const float* Wk_nope  = (const float*)d_in[7];
  const float* bk_nope  = (const float*)d_in[8];
  const float* Wk_rope  = (const float*)d_in[9];
  const float* bk_rope  = (const float*)d_in[10];
  const float* Wv_up    = (const float*)d_in[11];
  const float* bv_up    = (const float*)d_in[12];
  const float* Wo       = (const float*)d_in[13];
  const float* bo       = (const float*)d_in[14];

  char* ws = (char*)d_ws;
  size_t off = 0;
  auto alloc = [&](size_t bytes) -> __bf16* {
    __bf16* p = (__bf16*)(ws + off);
    off += (bytes + 255) & ~(size_t)255;
    return p;
  };

  __bf16* hsb   = alloc((size_t)MM * HH * 2);      // hidden bf16
  __bf16* wqd   = alloc((size_t)LL * HH * 2);
  __bf16* wkvd  = alloc((size_t)LL * HH * 2);
  __bf16* wqu   = alloc((size_t)HH * LL * 2);
  __bf16* wkn   = alloc((size_t)(HH / 2) * LL * 2);
  __bf16* wkr   = alloc((size_t)(HH / 2) * LL * 2);
  __bf16* wvu   = alloc((size_t)HH * LL * 2);
  __bf16* wob   = alloc((size_t)HH * HH * 2);
  __bf16* qlat  = alloc((size_t)MM * LL * 2);
  __bf16* kvlat = alloc((size_t)MM * LL * 2);
  __bf16* qm    = alloc((size_t)MM * HH * 2);
  __bf16* km    = alloc((size_t)MM * HH * 2);
  __bf16* krr   = alloc((size_t)MM * (HH / 2) * 2);
  __bf16* vt    = alloc((size_t)MM * HH * 2);      // V transposed per (b,h): [d][s]
  __bf16* aout  = alloc((size_t)MM * HH * 2);
  (void)ws_size;

  auto cvt = [&](const float* src, __bf16* dst, size_t n) {
    int blocks = (int)((n + 1023) / 1024);
    f32_to_bf16<<<blocks, 256, 0, stream>>>(src, dst, (int)n);
  };
  cvt(hs, hsb, (size_t)MM * HH);
  cvt(Wq_down, wqd, (size_t)LL * HH);
  cvt(Wkv_down, wkvd, (size_t)LL * HH);
  cvt(Wq_up, wqu, (size_t)HH * LL);
  cvt(Wk_nope, wkn, (size_t)(HH / 2) * LL);
  cvt(Wk_rope, wkr, (size_t)(HH / 2) * LL);
  cvt(Wv_up, wvu, (size_t)HH * LL);
  cvt(Wo, wob, (size_t)HH * HH);

  auto gemm = [&](const __bf16* X, const __bf16* Wb, const float* bias, void* Y,
                  int M, int N, int K, int ldy, int col0, int mode) {
    dim3 grid(N / 64, M / 256);
    wmma_gemm_xwt<<<grid, dim3(256), 0, stream>>>(X, Wb, bias, Y, M, N, K, ldy, col0, mode);
  };

  // 1) latents
  gemm(hsb, wqd, bq_down, qlat, MM, LL, HH, LL, 0, 0);
  gemm(hsb, wkvd, bkv_down, kvlat, MM, LL, HH, LL, 0, 0);
  // 2) q / k_nope (into km cols [0,1024)) / k_rope_raw / v (transposed)
  gemm(qlat, wqu, bq_up, qm, MM, HH, LL, HH, 0, 0);
  gemm(kvlat, wkn, bk_nope, km, MM, HH / 2, LL, HH, 0, 0);
  gemm(kvlat, wkr, bk_rope, krr, MM, HH / 2, LL, HH / 2, 0, 0);
  rope_kernel<<<(MM * 512) / 256, 256, 0, stream>>>(krr, km);
  gemm(kvlat, wvu, bv_up, vt, MM, HH, LL, 0, 0, 2);
  // 3) attention
  mla_flash_attn<<<512, 256, 0, stream>>>(qm, km, vt, aout);
  // 4) output projection (fp32 result)
  gemm(aout, wob, bo, d_out, MM, HH, HH, HH, 0, 1);
}
